// RecurrentAutoencoder_65000035057904
// MI455X (gfx1250) — matrix operand order repaired and checked by reading, not compiled, not run on hardware
//
#include <hip/hip_runtime.h>
#include <hip/hip_bf16.h>

// ---------------------------------------------------------------------------
// CDNA5 (gfx1250) recurrent autoencoder.
//   Phase 1: batched LSTM (B=4096,T=140,H=128) -- bf16 WMMA GEMM per step.
//            256 blocks x 128 thr (4 waves, 1 wave/SIMD). Whh fragments are
//            hoisted into ~256 VGPRs per wave before the time loop, so the
//            hot loop is: 4x A-frag LDS loads + 32 register-resident WMMAs.
//   Phase 2a: gx = h1 @ Wih_e2^T + biases  (WMMA GEMM, M=4096,K=128,N=256)
//   Phase 2b: serial LSTM (4096 steps, hidden 64), weights in registers.
//   Phase 3+4: serial decoder LSTMs (140 steps), fused, writes recon[140].
// ---------------------------------------------------------------------------

typedef __attribute__((ext_vector_type(16))) __bf16 v16bf;
typedef __attribute__((ext_vector_type(8)))  float  v8f;

union FragBF {
    v16bf v;
    uint4 q[2];
};

__device__ __forceinline__ unsigned short f2bf(float f) {
    unsigned int u = __float_as_uint(f);
    unsigned int r = (u + 0x7FFFu + ((u >> 16) & 1u)) >> 16;
    return (unsigned short)r;
}
__device__ __forceinline__ float sigf(float x) { return 1.0f / (1.0f + __expf(-x)); }
__device__ __forceinline__ float tanh_fast(float x) {
    x = fminf(fmaxf(x, -15.0f), 15.0f);
    float e = __expf(2.0f * x);
    return (e - 1.0f) / (e + 1.0f);
}

// ======================= Phase 1: encoder LSTM-1 ===========================
// grid = 256 blocks (16 batch rows each), 128 threads = 4 waves.
// Wave cb: all 16 rows, hidden cols cb*32..+32, all 4 gates.
#define HPAD 136   // padded row length (bf16 elems) -> bank-conflict free

__global__ __launch_bounds__(128)
void lstm_e1_wmma(const float* __restrict__ x,      // [4096][1][140]
                  const float* __restrict__ Wih,    // [512][1]
                  const float* __restrict__ Whh,    // [512][128]
                  const float* __restrict__ bih,    // [512]
                  const float* __restrict__ bhh,    // [512]
                  float* __restrict__ h1_out)       // [4096][128]
{
    extern __shared__ __align__(16) char smem[];
    unsigned short* Wl     = (unsigned short*)smem;           // [512][HPAD] bf16
    unsigned short* Hl     = Wl + 512 * HPAD;                 // [16][HPAD]  bf16
    float*          xs     = (float*)(Hl + 16 * HPAD);        // [16][140]
    float*          bias_s = xs + 16 * 140;                   // [512]
    float*          wih_s  = bias_s + 512;                    // [512]

    const int tid = threadIdx.x;
    const int b0  = blockIdx.x * 16;

    for (int idx = tid; idx < 512 * 128; idx += 128) {
        int r = idx >> 7, k = idx & 127;
        Wl[r * HPAD + k] = f2bf(Whh[idx]);
    }
    for (int g = tid; g < 512; g += 128) {
        bias_s[g] = bih[g] + bhh[g];
        wih_s[g]  = Wih[g];
    }
    for (int idx = tid; idx < 16 * 140; idx += 128) {
        int m = idx / 140, t = idx - m * 140;
        xs[m * 140 + t] = x[(size_t)(b0 + m) * 140 + t];
    }
    for (int idx = tid; idx < 16 * HPAD; idx += 128) Hl[idx] = 0;   // h0 = 0
    __syncthreads();

    const int wv   = tid >> 5;
    const int lane = tid & 31;
    const int cb   = wv;               // 0..3 : 32-col block of hidden
    const int ln   = lane & 15;
    const int lh   = lane >> 4;        // lane half

    // ---- hoist all B fragments (Whh) into registers: timestep-invariant ----
    FragBF Breg[4][4][2];              // [kstep][gate][ntile]
    #pragma unroll
    for (int ks = 0; ks < 4; ++ks) {
        #pragma unroll
        for (int gi = 0; gi < 4; ++gi) {
            #pragma unroll
            for (int nt = 0; nt < 2; ++nt) {
                const int col = gi * 128 + cb * 32 + nt * 16 + ln;
                const int kb  = ks * 32 + lh * 16;
                const uint4* bp = (const uint4*)&Wl[col * HPAD + kb];
                Breg[ks][gi][nt].q[0] = bp[0];
                Breg[ks][gi][nt].q[1] = bp[1];
            }
        }
    }

    // per-thread t-invariant activation constants
    float wi_c[2][4], bb_c[2][4];
    #pragma unroll
    for (int nt = 0; nt < 2; ++nt) {
        const int j = cb * 32 + nt * 16 + ln;
        #pragma unroll
        for (int gi = 0; gi < 4; ++gi) {
            wi_c[nt][gi] = wih_s[gi * 128 + j];
            bb_c[nt][gi] = bias_s[gi * 128 + j];
        }
    }

    float cst[2][8];                   // cell state (2 n-tiles x 8 rows/lane)
    #pragma unroll
    for (int nt = 0; nt < 2; ++nt)
        #pragma unroll
        for (int e = 0; e < 8; ++e) cst[nt][e] = 0.0f;

    const unsigned short* arow = &Hl[ln * HPAD];

    for (int t = 0; t < 140; ++t) {
        v8f acc[4][2];
        #pragma unroll
        for (int gi = 0; gi < 4; ++gi)
            #pragma unroll
            for (int nt = 0; nt < 2; ++nt) { v8f z = {}; acc[gi][nt] = z; }

        #pragma unroll
        for (int ks = 0; ks < 4; ++ks) {
            const int k0 = ks * 32;
            FragBF A;
            A.q[0] = *(const uint4*)&arow[k0 + lh * 16];
            A.q[1] = *(const uint4*)&arow[k0 + lh * 16 + 8];
            #pragma unroll
            for (int gi = 0; gi < 4; ++gi) {
                #pragma unroll
                for (int nt = 0; nt < 2; ++nt) {
                    acc[gi][nt] = __builtin_amdgcn_wmma_f32_16x16x32_bf16(
                        false, A.v, false, Breg[ks][gi][nt].v, (short)0,
                        acc[gi][nt], false, false);
                }
            }
        }
        __syncthreads();                       // GEMM reads of Hl done

        #pragma unroll
        for (int e = 0; e < 8; ++e) {
            const int m  = e + lh * 8;
            const float xv = xs[m * 140 + t];
            #pragma unroll
            for (int nt = 0; nt < 2; ++nt) {
                const int j = cb * 32 + nt * 16 + ln;
                float pi = acc[0][nt][e] + xv * wi_c[nt][0] + bb_c[nt][0];
                float pf = acc[1][nt][e] + xv * wi_c[nt][1] + bb_c[nt][1];
                float pg = acc[2][nt][e] + xv * wi_c[nt][2] + bb_c[nt][2];
                float po = acc[3][nt][e] + xv * wi_c[nt][3] + bb_c[nt][3];
                float c  = sigf(pf) * cst[nt][e] + sigf(pi) * tanh_fast(pg);
                cst[nt][e] = c;
                float h  = sigf(po) * tanh_fast(c);
                Hl[m * HPAD + j] = f2bf(h);
                if (t == 139) h1_out[(size_t)(b0 + m) * 128 + j] = h;
            }
        }
        __syncthreads();                       // Hl updated for next step
    }
}

// =============== Phase 2a: gx = h1 @ Wih_e2^T + (bih+bhh) ==================
// grid = 128 blocks (32 rows each), 256 threads; wave tile 16x64 (4 n-tiles).
__global__ __launch_bounds__(256)
void gx_e2_wmma(const float* __restrict__ h1,     // [4096][128]
                const float* __restrict__ Wih2,   // [256][128]
                const float* __restrict__ bih2,
                const float* __restrict__ bhh2,
                float* __restrict__ gx)           // [4096][256]
{
    extern __shared__ __align__(16) char smem[];
    unsigned short* Wl = (unsigned short*)smem;          // [256][HPAD]
    unsigned short* Al = Wl + 256 * HPAD;                // [32][HPAD]
    float*          bs = (float*)(Al + 32 * HPAD);       // [256]

    const int tid = threadIdx.x;
    const int b0  = blockIdx.x * 32;

    for (int idx = tid; idx < 256 * 128; idx += 256) {
        int r = idx >> 7, k = idx & 127;
        Wl[r * HPAD + k] = f2bf(Wih2[idx]);
    }
    for (int idx = tid; idx < 32 * 128; idx += 256) {
        int m = idx >> 7, k = idx & 127;
        Al[m * HPAD + k] = f2bf(h1[(size_t)(b0 + m) * 128 + k]);
    }
    for (int g = tid; g < 256; g += 256) bs[g] = bih2[g] + bhh2[g];
    __syncthreads();

    const int wv   = tid >> 5;
    const int lane = tid & 31;
    const int rb   = wv >> 2;
    const int cb   = wv & 3;          // 64-col block
    const int ln   = lane & 15;
    const int lh   = lane >> 4;

    v8f acc[4];
    #pragma unroll
    for (int nt = 0; nt < 4; ++nt) { v8f z = {}; acc[nt] = z; }

    const unsigned short* arow = &Al[(rb * 16 + ln) * HPAD];
    #pragma unroll
    for (int ks = 0; ks < 4; ++ks) {
        const int k0 = ks * 32;
        FragBF A;
        A.q[0] = *(const uint4*)&arow[k0 + lh * 16];
        A.q[1] = *(const uint4*)&arow[k0 + lh * 16 + 8];
        #pragma unroll
        for (int nt = 0; nt < 4; ++nt) {
            const int col = cb * 64 + nt * 16 + ln;
            const int kb  = k0 + lh * 16;
            FragBF B;
            const uint4* bp = (const uint4*)&Wl[col * HPAD + kb];
            B.q[0] = bp[0];
            B.q[1] = bp[1];
            acc[nt] = __builtin_amdgcn_wmma_f32_16x16x32_bf16(
                false, A.v, false, B.v, (short)0, acc[nt], false, false);
        }
    }
    #pragma unroll
    for (int nt = 0; nt < 4; ++nt) {
        const int n = cb * 64 + nt * 16 + ln;
        const float bn = bs[n];
        #pragma unroll
        for (int e = 0; e < 8; ++e) {
            const int m = rb * 16 + e + lh * 8;
            gx[(size_t)(b0 + m) * 256 + n] = acc[nt][e] + bn;
        }
    }
}

// ================= Phase 2b: serial encoder LSTM-2 =========================
// 1 block, 256 threads. Whh_e2 rows in registers; gx prefetched 1 step ahead.
__global__ __launch_bounds__(256)
void lstm_e2_serial(const float* __restrict__ gx,      // [4096][256]
                    const float* __restrict__ Whh2,    // [256][64]
                    const float* __restrict__ Wih_d1,  // [512][64]
                    const float* __restrict__ bih_d1,
                    const float* __restrict__ bhh_d1,
                    float* __restrict__ gxd1)          // [512] out
{
    __shared__ float h_s[64];
    __shared__ float gate_s[256];
    const int tid = threadIdx.x;

    float w[64];
    #pragma unroll
    for (int k = 0; k < 64; ++k) w[k] = Whh2[tid * 64 + k];
    if (tid < 64) h_s[tid] = 0.0f;
    float c = 0.0f;
    __syncthreads();

    float gcur = gx[tid];
    for (int t = 0; t < 4096; ++t) {
        float gnext = (t < 4095) ? gx[(size_t)(t + 1) * 256 + tid] : 0.0f;
        float s = gcur;
        const float2* hr = (const float2*)h_s;
        #pragma unroll
        for (int kk = 0; kk < 32; ++kk) {
            float2 h2 = hr[kk];
            s += w[2 * kk] * h2.x + w[2 * kk + 1] * h2.y;
        }
        gate_s[tid] = s;
        __syncthreads();
        if (tid < 64) {
            float pi = gate_s[tid], pf = gate_s[64 + tid];
            float pg = gate_s[128 + tid], po = gate_s[192 + tid];
            c = sigf(pf) * c + sigf(pi) * tanh_fast(pg);
            h_s[tid] = sigf(po) * tanh_fast(c);
        }
        __syncthreads();
        gcur = gnext;
    }
    // gxd1 = z @ Wih_d1^T + bih_d1 + bhh_d1   (z == h_s)
    for (int g = tid; g < 512; g += 256) {
        float s = bih_d1[g] + bhh_d1[g];
        #pragma unroll 8
        for (int k = 0; k < 64; ++k) s += Wih_d1[g * 64 + k] * h_s[k];
        gxd1[g] = s;
    }
}

// ============ Phase 3+4: decoder LSTM-1 (H=128) + LSTM-2 (H=1) =============
#define DPAD 130   // fp32 row pad: float2-aligned, conflict-free

__global__ __launch_bounds__(512)
void lstm_d1_d2(const float* __restrict__ Whh_d1,   // [512][128]
                const float* __restrict__ gxd1,     // [512]
                const float* __restrict__ Wih_d2,   // [4][128]
                const float* __restrict__ Whh_d2,   // [4][1]
                const float* __restrict__ bih_d2,   // [4]
                const float* __restrict__ bhh_d2,   // [4]
                float* __restrict__ out)            // [140]
{
    extern __shared__ __align__(16) char smem[];
    float* Wl     = (float*)smem;         // [512][DPAD]
    float* h_s    = Wl + 512 * DPAD;      // [128]
    float* gate_s = h_s + 128;            // [512]
    float* wd2_s  = gate_s + 512;         // [512]
    __shared__ float pre4[4];

    const int tid = threadIdx.x;
    for (int idx = tid; idx < 512 * 128; idx += 512) {
        int g = idx >> 7, k = idx & 127;
        Wl[g * DPAD + k] = Whh_d1[idx];
    }
    wd2_s[tid] = Wih_d2[tid];
    if (tid < 128) h_s[tid] = 0.0f;
    const float gx0 = gxd1[tid];
    float c = 0.0f, h2 = 0.0f, c2 = 0.0f;
    __syncthreads();

    for (int t = 0; t < 140; ++t) {
        float s = gx0;
        const float2* wr = (const float2*)&Wl[tid * DPAD];
        const float2* hr = (const float2*)h_s;
        #pragma unroll
        for (int kk = 0; kk < 64; ++kk) {
            float2 a = wr[kk], b = hr[kk];
            s += a.x * b.x + a.y * b.y;
        }
        gate_s[tid] = s;
        __syncthreads();
        if (tid < 128) {
            float pi = gate_s[tid], pf = gate_s[128 + tid];
            float pg = gate_s[256 + tid], po = gate_s[384 + tid];
            c = sigf(pf) * c + sigf(pi) * tanh_fast(pg);
            h_s[tid] = sigf(po) * tanh_fast(c);
        }
        __syncthreads();
        if (tid < 4) {
            float s2 = 0.0f;
            #pragma unroll 8
            for (int k = 0; k < 128; ++k) s2 += wd2_s[tid * 128 + k] * h_s[k];
            pre4[tid] = s2;
        }
        __syncthreads();
        if (tid == 0) {
            float pi = pre4[0] + bih_d2[0] + bhh_d2[0] + h2 * Whh_d2[0];
            float pf = pre4[1] + bih_d2[1] + bhh_d2[1] + h2 * Whh_d2[1];
            float pg = pre4[2] + bih_d2[2] + bhh_d2[2] + h2 * Whh_d2[2];
            float po = pre4[3] + bih_d2[3] + bhh_d2[3] + h2 * Whh_d2[3];
            c2 = sigf(pf) * c2 + sigf(pi) * tanh_fast(pg);
            h2 = sigf(po) * tanh_fast(c2);
            out[t] = h2;
        }
        __syncthreads();
    }
}

// ============================== launcher ===================================
extern "C" void kernel_launch(void* const* d_in, const int* in_sizes, int n_in,
                              void* d_out, int out_size, void* d_ws, size_t ws_size,
                              hipStream_t stream) {
    const float* x      = (const float*)d_in[0];
    const float* Wih_e1 = (const float*)d_in[1];
    const float* Whh_e1 = (const float*)d_in[2];
    const float* bih_e1 = (const float*)d_in[3];
    const float* bhh_e1 = (const float*)d_in[4];
    const float* Wih_e2 = (const float*)d_in[5];
    const float* Whh_e2 = (const float*)d_in[6];
    const float* bih_e2 = (const float*)d_in[7];
    const float* bhh_e2 = (const float*)d_in[8];
    const float* Wih_d1 = (const float*)d_in[9];
    const float* Whh_d1 = (const float*)d_in[10];
    const float* bih_d1 = (const float*)d_in[11];
    const float* bhh_d1 = (const float*)d_in[12];
    const float* Wih_d2 = (const float*)d_in[13];
    const float* Whh_d2 = (const float*)d_in[14];
    const float* bih_d2 = (const float*)d_in[15];
    const float* bhh_d2 = (const float*)d_in[16];

    float* ws   = (float*)d_ws;
    float* h1   = ws;                         // 4096*128
    float* gx   = h1 + 4096 * 128;            // 4096*256
    float* gxd1 = gx + (size_t)4096 * 256;    // 512
    float* outp = (float*)d_out;              // 140

    constexpr size_t SH1 = 512 * HPAD * 2 + 16 * HPAD * 2 + 16 * 140 * 4 + 512 * 4 + 512 * 4;
    constexpr size_t SH2 = 256 * HPAD * 2 + 32 * HPAD * 2 + 256 * 4;
    constexpr size_t SH4 = (512 * DPAD + 128 + 512 + 512) * 4;

    lstm_e1_wmma<<<256, 128, SH1, stream>>>(x, Wih_e1, Whh_e1, bih_e1, bhh_e1, h1);
    gx_e2_wmma<<<128, 256, SH2, stream>>>(h1, Wih_e2, bih_e2, bhh_e2, gx);
    lstm_e2_serial<<<1, 256, 0, stream>>>(gx, Whh_e2, Wih_d1, bih_d1, bhh_d1, gxd1);
    lstm_d1_d2<<<1, 512, SH4, stream>>>(Whh_d1, gxd1, Wih_d2, Whh_d2, bih_d2, bhh_d2, outp);
}
